// DAEM_63015760167711
// MI455X (gfx1250) — matrix operand order repaired and checked by reading, not compiled, not run on hardware
//
#include <hip/hip_runtime.h>
#include <math.h>

// ---------------------------------------------------------------------------
// Types for CDNA5 WMMA (wave32, 16x16x32 bf16 -> f32)
// ---------------------------------------------------------------------------
typedef __attribute__((ext_vector_type(16))) __bf16 v16bf;
typedef __attribute__((ext_vector_type(8)))  __bf16 v8bf;
typedef __attribute__((ext_vector_type(8)))  float  v8f;

// ---------------------------------------------------------------------------
// Fragment loaders, matching ISA 7.12.2 VGPR layouts.
// A (16x32 bf16, row-major [M,K]): lane l<16 -> row m0+l, elems 0..7 = K k0..k0+7,
//   elems 8..15 = K k0+16..k0+23 ; lanes 16..31 same rows, K +8 / +24.
// B (32x16 bf16) fed from Bt row-major [N,K]: lane n<16 -> col n0+n holds
//   K k0..k0+7, k0+16..k0+23; lanes 16..31 -> K +8 / +24 (same as A).
// ---------------------------------------------------------------------------
__device__ __forceinline__ v16bf load_a_frag(const __bf16* __restrict__ A, int lda,
                                             int m0, int k0, int lane) {
  int row = m0 + (lane & 15);
  int kb  = k0 + ((lane & 16) ? 8 : 0);
  const __bf16* p = A + (size_t)row * lda + kb;
  v8bf lo = *(const v8bf*)(p);        // 16B
  v8bf hi = *(const v8bf*)(p + 16);   // 16B, +32B offset
  v16bf r;
#pragma unroll
  for (int i = 0; i < 8; ++i) { r[i] = lo[i]; r[i + 8] = hi[i]; }
  return r;
}

__device__ __forceinline__ v16bf load_b_frag(const __bf16* __restrict__ Bt, int ldbt,
                                             int n0, int k0, int lane) {
  int row = n0 + (lane & 15);
  int kb  = k0 + ((lane & 16) ? 8 : 0);
  const __bf16* p = Bt + (size_t)row * ldbt + kb;
  v8bf lo = *(const v8bf*)(p);        // 16B
  v8bf hi = *(const v8bf*)(p + 16);   // 16B, +32B offset
  v16bf r;
#pragma unroll
  for (int i = 0; i < 8; ++i) { r[i] = lo[i]; r[i + 8] = hi[i]; }
  return r;
}

// ---------------------------------------------------------------------------
// Generic batched GEMM: C[M,N] = A[M,K](bf16) * B[K,N] + bias,
// with B given as Bt[N,K] (bf16, row-major). f32 accumulation via WMMA.
// Optional outputs: f32 C (ldc), bf16 C (ldc), bf16 Ct[N,M] (ldct).
// Block = 256 threads = 8 waves, tile 64(M) x 128(N); each wave 32x32 (2x2 WMMA).
// All M,N,K used here divide 64/128/32 exactly.
// ---------------------------------------------------------------------------
__global__ void __launch_bounds__(256)
gemm_bf16_wmma(const __bf16* __restrict__ A, size_t strideA, int lda,
               const __bf16* __restrict__ Bt, size_t strideBt, int ldbt,
               const float* __restrict__ bias,
               float* __restrict__ Cf, size_t strideCf, int ldc,
               __bf16* __restrict__ Cbf, size_t strideCbf,
               __bf16* __restrict__ Ctbf, size_t strideCt, int ldct,
               int M, int N, int K) {
  const int b    = blockIdx.z;
  const int lane = threadIdx.x & 31;
  const int wid  = threadIdx.x >> 5;
  const int mBase = blockIdx.y * 64  + (wid >> 2) * 32;
  const int nBase = blockIdx.x * 128 + (wid & 3) * 32;
  if (mBase >= M || nBase >= N) return;

  const __bf16* Ab  = A  + (size_t)b * strideA;
  const __bf16* Btb = Bt + (size_t)b * strideBt;

  v8f acc00 = {}, acc01 = {}, acc10 = {}, acc11 = {};

  for (int k0 = 0; k0 < K; k0 += 32) {
    if (k0 + 32 < K) {  // warm L2/L0 for the next K-tile (global_prefetch_b8)
      __builtin_prefetch(Ab  + (size_t)(mBase + (lane & 15)) * lda  + k0 + 32, 0, 1);
      __builtin_prefetch(Btb + (size_t)(nBase + (lane & 15)) * ldbt + k0 + 32, 0, 1);
    }
    v16bf a0 = load_a_frag(Ab, lda, mBase,      k0, lane);
    v16bf a1 = load_a_frag(Ab, lda, mBase + 16, k0, lane);
    v16bf b0 = load_b_frag(Btb, ldbt, nBase,      k0, lane);
    v16bf b1 = load_b_frag(Btb, ldbt, nBase + 16, k0, lane);
    acc00 = __builtin_amdgcn_wmma_f32_16x16x32_bf16(false, a0, false, b0, (short)0, acc00, false, false);
    acc01 = __builtin_amdgcn_wmma_f32_16x16x32_bf16(false, a0, false, b1, (short)0, acc01, false, false);
    acc10 = __builtin_amdgcn_wmma_f32_16x16x32_bf16(false, a1, false, b0, (short)0, acc10, false, false);
    acc11 = __builtin_amdgcn_wmma_f32_16x16x32_bf16(false, a1, false, b1, (short)0, acc11, false, false);
  }

  // C/D 16x16 f32 layout: VGPR v, lane<16 -> (M=v, N=lane); lane>=16 -> (M=v+8, N=lane-16)
  const int colOff = lane & 15;
  const int rowSel = (lane >> 4) * 8;
  v8f accs[2][2] = {{acc00, acc01}, {acc10, acc11}};
#pragma unroll
  for (int im = 0; im < 2; ++im)
#pragma unroll
    for (int in = 0; in < 2; ++in) {
      int mT = mBase + 16 * im, nT = nBase + 16 * in;
#pragma unroll
      for (int v = 0; v < 8; ++v) {
        int m = mT + v + rowSel;
        int n = nT + colOff;
        float val = accs[im][in][v];
        if (bias) val += bias[m];
        if (Cf)   Cf  [(size_t)b * strideCf  + (size_t)m * ldc  + n] = val;
        if (Cbf)  Cbf [(size_t)b * strideCbf + (size_t)m * ldc  + n] = (__bf16)val;
        if (Ctbf) Ctbf[(size_t)b * strideCt  + (size_t)n * ldct + m] = (__bf16)val;
      }
    }
}

// ---------------------------------------------------------------------------
// fp32 -> bf16 weight convert
// ---------------------------------------------------------------------------
__global__ void f32_to_bf16_kernel(const float* __restrict__ s, __bf16* __restrict__ d, int n) {
  int i = blockIdx.x * 256 + threadIdx.x;
  if (i < n) d[i] = (__bf16)s[i];
}

// ---------------------------------------------------------------------------
// Antialiased triangle-filter resize (s x s -> 32 x 32, scale = s/32 integer),
// fused with sigmoid channel-attention mask; writes masked features TRANSPOSED
// as bf16 into mfamT[b][p][choff + c] (ld = 960). One block per (b, out-pixel).
// scale==1 degenerates to identity (level 3).
// ---------------------------------------------------------------------------
__global__ void __launch_bounds__(256)
resize_mask_kernel(const float* __restrict__ f, int C, int s, int scale,
                   const float* __restrict__ attw, const float* __restrict__ attb,
                   __bf16* __restrict__ outT, int choff) {
  const int p = blockIdx.x;          // 0..1023
  const int b = blockIdx.y;
  const int oy = p >> 5, ox = p & 31;
  const int t = threadIdx.x;

  __shared__ float wy[17], wx[17];
  __shared__ float red[256];
  __shared__ float wnorm, a_s;

  const int ntaps = 2 * scale + 1;   // <= 17
  const float cy = (oy + 0.5f) * (float)scale - 0.5f;
  const float cx = (ox + 0.5f) * (float)scale - 0.5f;
  const int y0 = (int)ceilf(cy - (float)scale);
  const int x0 = (int)ceilf(cx - (float)scale);
  if (t < ntaps) {
    wy[t] = fmaxf(0.f, 1.f - fabsf((float)(y0 + t) - cy) / (float)scale);
    wx[t] = fmaxf(0.f, 1.f - fabsf((float)(x0 + t) - cx) / (float)scale);
  }
  __syncthreads();
  if (t == 0) {
    float sy = 0.f, sx = 0.f;
    for (int i = 0; i < ntaps; ++i) {
      int y = y0 + i; if (y >= 0 && y < s) sy += wy[i];
      int x = x0 + i; if (x >= 0 && x < s) sx += wx[i];
    }
    wnorm = 1.f / fmaxf(sy * sx, 1e-20f);
  }
  __syncthreads();

  float g[2] = {0.f, 0.f};
  const int nper = (C + 255) / 256;  // 1 or 2
  float partial = 0.f;
  for (int j = 0; j < nper; ++j) {
    int c = t + j * 256;
    float acc = 0.f;
    if (c < C) {
      const float* fp = f + ((size_t)b * C + c) * (size_t)s * s;
      for (int iy = 0; iy < ntaps; ++iy) {
        int y = y0 + iy;
        if ((unsigned)y >= (unsigned)s) continue;
        float wyv = wy[iy];
        if (wyv == 0.f) continue;
        const float* row = fp + (size_t)y * s;
        for (int ix = 0; ix < ntaps; ++ix) {
          int x = x0 + ix;
          if ((unsigned)x >= (unsigned)s) continue;
          acc += wyv * wx[ix] * row[x];
        }
      }
      acc *= wnorm;
      partial += attw[c] * acc;
    }
    g[j] = acc;
  }

  red[t] = partial;
  __syncthreads();
  for (int o = 128; o > 0; o >>= 1) {
    if (t < o) red[t] += red[t + o];
    __syncthreads();
  }
  if (t == 0) a_s = 1.f / (1.f + expf(-(red[0] + attb[0])));
  __syncthreads();
  const float a = a_s;

  for (int j = 0; j < nper; ++j) {
    int c = t + j * 256;
    if (c < C)
      outT[((size_t)b * 1024 + p) * 960 + choff + c] = (__bf16)(g[j] * a);
  }
}

// ---------------------------------------------------------------------------
// 7x7 count-valid box mean/var normalization over a 32x32 plane, optional
// exact GELU, writes f32 [b][c][p] and/or bf16 transposed [b][p][choff+c].
// One block per (channel, batch); plane cached in LDS.
// ---------------------------------------------------------------------------
__global__ void __launch_bounds__(256)
patch_norm_kernel(const float* __restrict__ X, int C,
                  float* __restrict__ outF, int gelu,
                  __bf16* __restrict__ outT, int ldt, int choff) {
  const int c = blockIdx.x, b = blockIdx.y;
  const float* xp = X + ((size_t)b * C + c) * 1024;
  __shared__ float sx[1024];
  const int t = threadIdx.x;
  for (int i = t; i < 1024; i += 256) sx[i] = xp[i];
  __syncthreads();
  for (int i = t; i < 1024; i += 256) {
    int y = i >> 5, x = i & 31;
    int yl = y - 3; if (yl < 0) yl = 0;
    int yh = y + 3; if (yh > 31) yh = 31;
    int xl = x - 3; if (xl < 0) xl = 0;
    int xh = x + 3; if (xh > 31) xh = 31;
    float sum = 0.f, sq = 0.f;
    for (int yy = yl; yy <= yh; ++yy) {
      const float* r = sx + (yy << 5);
      for (int xx = xl; xx <= xh; ++xx) { float v = r[xx]; sum += v; sq += v * v; }
    }
    float cnt = (float)((yh - yl + 1) * (xh - xl + 1));
    float mu  = sum / cnt;
    float var = fmaxf(sq / cnt - mu * mu, 0.f);
    float yv  = (sx[i] - mu) * rsqrtf(var + 1e-5f);
    if (gelu) yv = 0.5f * yv * (1.f + erff(yv * 0.70710678118f));
    if (outF) outF[((size_t)b * C + c) * 1024 + i] = yv;
    if (outT) outT[((size_t)b * 1024 + i) * ldt + choff + c] = (__bf16)yv;
  }
}

// ---------------------------------------------------------------------------
// Row softmax of logits [b][1024][1024] f32 -> bf16 (row-major = Bt for V*A^T)
// ---------------------------------------------------------------------------
__global__ void __launch_bounds__(256)
softmax_kernel(const float* __restrict__ L, __bf16* __restrict__ Aout) {
  const int i = blockIdx.x, b = blockIdx.y;
  const float* row = L + ((size_t)b * 1024 + i) * 1024;
  __shared__ float red[256];
  const int t = threadIdx.x;
  float v0 = row[t], v1 = row[t + 256], v2 = row[t + 512], v3 = row[t + 768];
  float m = fmaxf(fmaxf(v0, v1), fmaxf(v2, v3));
  red[t] = m; __syncthreads();
  for (int o = 128; o > 0; o >>= 1) { if (t < o) red[t] = fmaxf(red[t], red[t + o]); __syncthreads(); }
  m = red[0]; __syncthreads();
  v0 = expf(v0 - m); v1 = expf(v1 - m); v2 = expf(v2 - m); v3 = expf(v3 - m);
  red[t] = v0 + v1 + v2 + v3; __syncthreads();
  for (int o = 128; o > 0; o >>= 1) { if (t < o) red[t] += red[t + o]; __syncthreads(); }
  float inv = 1.f / red[0];
  __bf16* o = Aout + ((size_t)b * 1024 + i) * 1024;
  o[t] = (__bf16)(v0 * inv); o[t + 256] = (__bf16)(v1 * inv);
  o[t + 512] = (__bf16)(v2 * inv); o[t + 768] = (__bf16)(v3 * inv);
}

// ---------------------------------------------------------------------------
// S = M * fxn (elementwise, f32), written transposed as bf16 [b][p][c] (ld=512)
// ---------------------------------------------------------------------------
__global__ void smul_transpose_kernel(const float* __restrict__ Mb,
                                      const float* __restrict__ Fn,
                                      __bf16* __restrict__ St) {
  size_t i = (size_t)blockIdx.x * 256 + threadIdx.x;   // over 8*512*1024
  if (i >= (size_t)8 * 512 * 1024) return;
  int p = (int)(i & 1023);
  size_t bc = i >> 10;
  int c = (int)(bc & 511);
  int b = (int)(bc >> 9);
  St[((size_t)b * 1024 + p) * 512 + c] = (__bf16)(Mb[i] * Fn[i]);
}

// ---------------------------------------------------------------------------
// fp32 [b][C][1024] -> bf16 transposed [b][1024][C]
// ---------------------------------------------------------------------------
__global__ void transpose_bf16_kernel(const float* __restrict__ X,
                                      __bf16* __restrict__ Xt, int C) {
  size_t i = (size_t)blockIdx.x * 256 + threadIdx.x;
  if (i >= (size_t)8 * C * 1024) return;
  int p = (int)(i & 1023);
  size_t bc = i >> 10;
  int c = (int)(bc % C);
  int b = (int)(bc / C);
  Xt[((size_t)b * 1024 + p) * C + c] = (__bf16)X[i];
}

// ---------------------------------------------------------------------------
// Host driver
// ---------------------------------------------------------------------------
extern "C" void kernel_launch(void* const* d_in, const int* in_sizes, int n_in,
                              void* d_out, int out_size, void* d_ws, size_t ws_size,
                              hipStream_t stream) {
  (void)in_sizes; (void)n_in; (void)out_size; (void)ws_size;
  const int B = 8;
  const int CHS[4] = {64, 128, 256, 512};
  const int CHOFF[4] = {0, 64, 192, 448};

  // Input order = setup_inputs() dict order
  const float* ff[4] = {(const float*)d_in[0], (const float*)d_in[2],
                        (const float*)d_in[4], (const float*)d_in[6]};
  const float* bfp[4] = {(const float*)d_in[1], (const float*)d_in[3],
                         (const float*)d_in[5], (const float*)d_in[7]};
  const float* attw[2][4]; const float* attb[2][4];
  const float* projw[2][4]; const float* projb[2][4];
  for (int i = 0; i < 4; ++i) {
    attw[0][i]  = (const float*)d_in[8 + i];
    attb[0][i]  = (const float*)d_in[12 + i];
    projw[0][i] = (const float*)d_in[16 + i];
    projb[0][i] = (const float*)d_in[20 + i];
    attw[1][i]  = (const float*)d_in[24 + i];
    attb[1][i]  = (const float*)d_in[28 + i];
    projw[1][i] = (const float*)d_in[32 + i];
    projb[1][i] = (const float*)d_in[36 + i];
  }
  const float* c1w = (const float*)d_in[40]; const float* c1b = (const float*)d_in[41];
  const float* c2w = (const float*)d_in[42]; const float* c2b = (const float*)d_in[43];
  const float* c3w = (const float*)d_in[44]; const float* c3b = (const float*)d_in[45];
  const float* c4w = (const float*)d_in[46]; const float* c4b = (const float*)d_in[47];
  const float* c5w = (const float*)d_in[48]; const float* c5b = (const float*)d_in[49];

  // Bump allocator over d_ws (256B aligned)
  char* base = (char*)d_ws;
  size_t off = 0;
  auto alloc = [&](size_t bytes) -> void* {
    off = (off + 255) & ~(size_t)255;
    void* p = base + off;
    off += bytes;
    return p;
  };

  // bf16 weights
  __bf16* projw_bf[2][4];
  for (int pr = 0; pr < 2; ++pr)
    for (int i = 0; i < 4; ++i)
      projw_bf[pr][i] = (__bf16*)alloc((size_t)CHS[i] * CHS[i] * 2);
  __bf16* c1w_bf = (__bf16*)alloc((size_t)960 * 960 * 2);
  __bf16* c2w_bf = (__bf16*)alloc((size_t)960 * 960 * 2);
  __bf16* c3w_bf = (__bf16*)alloc((size_t)512 * 512 * 2);
  __bf16* c4w_bf = (__bf16*)alloc((size_t)512 * 512 * 2);
  __bf16* c5w_bf = (__bf16*)alloc((size_t)512 * 512 * 2);

  // Intermediates
  __bf16* mfamT[2]; float* mfamF[2]; __bf16* normT[2];
  for (int pr = 0; pr < 2; ++pr) {
    mfamT[pr] = (__bf16*)alloc((size_t)B * 1024 * 960 * 2);  // masked feats, transposed bf16
    mfamF[pr] = (float*)alloc((size_t)B * 960 * 1024 * 4);   // proj outputs f32
    normT[pr] = (__bf16*)alloc((size_t)B * 1024 * 960 * 2);  // f1x/b1x transposed bf16
  }
  __bf16* QT    = (__bf16*)alloc((size_t)B * 1024 * 960 * 2);
  __bf16* KT    = (__bf16*)alloc((size_t)B * 1024 * 960 * 2);
  __bf16* fxnT  = (__bf16*)alloc((size_t)B * 1024 * 512 * 2); // norm(ff3)^T bf16
  __bf16* bxT   = (__bf16*)alloc((size_t)B * 1024 * 512 * 2); // bf3^T bf16
  __bf16* Vbf   = (__bf16*)alloc((size_t)B * 512 * 1024 * 2);
  float*  fxn   = (float*)alloc((size_t)B * 512 * 1024 * 4);  // c4 output f32
  float*  logits= (float*)alloc((size_t)B * 1024 * 1024 * 4);
  __bf16* Asm   = (__bf16*)alloc((size_t)B * 1024 * 1024 * 2);
  float*  Mbuf  = (float*)alloc((size_t)B * 512 * 1024 * 4);
  __bf16* St    = (__bf16*)alloc((size_t)B * 1024 * 512 * 2);
  float*  c5out = (float*)alloc((size_t)B * 512 * 1024 * 4);

  auto cvt = [&](const float* s, __bf16* d, int n) {
    f32_to_bf16_kernel<<<(n + 255) / 256, 256, 0, stream>>>(s, d, n);
  };
  for (int pr = 0; pr < 2; ++pr)
    for (int i = 0; i < 4; ++i)
      cvt(projw[pr][i], projw_bf[pr][i], CHS[i] * CHS[i]);
  cvt(c1w, c1w_bf, 960 * 960); cvt(c2w, c2w_bf, 960 * 960);
  cvt(c3w, c3w_bf, 512 * 512); cvt(c4w, c4w_bf, 512 * 512); cvt(c5w, c5w_bf, 512 * 512);

  // 1) resize + sigmoid-mask, transposed bf16
  for (int pr = 0; pr < 2; ++pr) {
    const float* const* src = (pr == 0) ? ff : bfp;
    for (int i = 0; i < 4; ++i) {
      int s = 256 >> i, scale = 8 >> i;
      resize_mask_kernel<<<dim3(1024, B), 256, 0, stream>>>(
          src[i], CHS[i], s, scale, attw[pr][i], attb[pr][i], mfamT[pr], CHOFF[i]);
    }
  }

  // 2) per-level proj GEMMs into concatenated mfam f32 [B,960,1024]
  for (int pr = 0; pr < 2; ++pr)
    for (int i = 0; i < 4; ++i) {
      int C = CHS[i];
      gemm_bf16_wmma<<<dim3(1024 / 128, C / 64, B), 256, 0, stream>>>(
          projw_bf[pr][i], 0, C,
          mfamT[pr] + CHOFF[i], (size_t)1024 * 960, 960,
          projb[pr][i],
          mfamF[pr] + (size_t)CHOFF[i] * 1024, (size_t)960 * 1024, 1024,
          nullptr, 0, nullptr, 0, 0,
          C, 1024, C);
    }

  // 3) patch_mv_norm -> transposed bf16 (GEMM inputs)
  for (int pr = 0; pr < 2; ++pr)
    patch_norm_kernel<<<dim3(960, B), 256, 0, stream>>>(
        mfamF[pr], 960, nullptr, 0, normT[pr], 960, 0);
  patch_norm_kernel<<<dim3(512, B), 256, 0, stream>>>(
      ff[3], 512, nullptr, 0, fxnT, 512, 0);
  transpose_bf16_kernel<<<(int)(((size_t)B * 512 * 1024 + 255) / 256), 256, 0, stream>>>(
      bfp[3], bxT, 512);

  // 4) Q,K (write transposed bf16), V (bf16 row-major), c4 (f32)
  gemm_bf16_wmma<<<dim3(8, 15, B), 256, 0, stream>>>(
      c1w_bf, 0, 960, normT[0], (size_t)1024 * 960, 960, c1b,
      nullptr, 0, 1024, nullptr, 0, QT, (size_t)1024 * 960, 960, 960, 1024, 960);
  gemm_bf16_wmma<<<dim3(8, 15, B), 256, 0, stream>>>(
      c2w_bf, 0, 960, normT[1], (size_t)1024 * 960, 960, c2b,
      nullptr, 0, 1024, nullptr, 0, KT, (size_t)1024 * 960, 960, 960, 1024, 960);
  gemm_bf16_wmma<<<dim3(8, 8, B), 256, 0, stream>>>(
      c3w_bf, 0, 512, bxT, (size_t)1024 * 512, 512, c3b,
      nullptr, 0, 1024, Vbf, (size_t)512 * 1024, nullptr, 0, 0, 512, 1024, 512);
  gemm_bf16_wmma<<<dim3(8, 8, B), 256, 0, stream>>>(
      c4w_bf, 0, 512, fxnT, (size_t)1024 * 512, 512, c4b,
      fxn, (size_t)512 * 1024, 1024, nullptr, 0, nullptr, 0, 0, 512, 1024, 512);

  // 5) logits = Q^T K : A = QT [1024,960], Bt = KT [1024,960]
  gemm_bf16_wmma<<<dim3(8, 16, B), 256, 0, stream>>>(
      QT, (size_t)1024 * 960, 960, KT, (size_t)1024 * 960, 960, nullptr,
      logits, (size_t)1024 * 1024, 1024, nullptr, 0, nullptr, 0, 0, 1024, 1024, 960);

  // 6) softmax rows -> bf16 (directly the Bt of the next GEMM)
  softmax_kernel<<<dim3(1024, B), 256, 0, stream>>>(logits, Asm);

  // 7) M = V A^T : A = V [512,1024], Bt = Asm [1024,1024]
  gemm_bf16_wmma<<<dim3(8, 8, B), 256, 0, stream>>>(
      Vbf, (size_t)512 * 1024, 1024, Asm, (size_t)1024 * 1024, 1024, nullptr,
      Mbuf, (size_t)512 * 1024, 1024, nullptr, 0, nullptr, 0, 0, 512, 1024, 1024);

  // 8) S = M * fxn -> S^T bf16
  smul_transpose_kernel<<<(int)(((size_t)B * 512 * 1024 + 255) / 256), 256, 0, stream>>>(
      Mbuf, fxn, St);

  // 9) c5 GEMM -> f32
  gemm_bf16_wmma<<<dim3(8, 8, B), 256, 0, stream>>>(
      c5w_bf, 0, 512, St, (size_t)1024 * 512, 512, c5b,
      c5out, (size_t)512 * 1024, 1024, nullptr, 0, nullptr, 0, 0, 512, 1024, 512);

  // 10) final patch_mv_norm + exact GELU -> d_out [8,512,32,32] f32
  patch_norm_kernel<<<dim3(512, B), 256, 0, stream>>>(
      c5out, 512, (float*)d_out, 1, nullptr, 0, 0);
}
